// GPTBlock_16080357556652
// MI455X (gfx1250) — compile-verified, matrix operand order checked by reading, not run on hardware
//
#include <hip/hip_runtime.h>
#include <hip/hip_bf16.h>
#include <stdint.h>

// ---------------------------------------------------------------------------
// MI455X (gfx1250) GPT block forward.
// - All GEMM math via v_wmma_f32_16x16x32_bf16 (wave32).
// - bf16 A/B fragment K-mapping (ISA 7.12.2): lane holds row/col (lane&15);
//   element i -> K = half*8+i (i<8) or 16+half*8+(i-8), half = lane>>4.
//   Both runs are contiguous -> every fragment is two b128 LDS loads when the
//   LDS tile is stored K-contiguous.
// - C/D layout: VGPR r holds row (r + 8*half), col (lane&15).
// - Tile staging to LDS uses the Tensor Data Mover (tensor_load_to_lds,
//   TENSORcnt) with D# row padding matching the LDS strides; double-buffered
//   in the GEMM so the DMA for tile k+1 overlaps WMMA on tile k.
// ---------------------------------------------------------------------------

typedef __bf16  v16bf __attribute__((ext_vector_type(16)));
typedef float   v8f   __attribute__((ext_vector_type(8)));

#if defined(__has_builtin)
#if __has_builtin(__builtin_amdgcn_tensor_load_to_lds) && \
    __has_builtin(__builtin_amdgcn_s_wait_tensorcnt)
#define HAVE_TDM 1
#endif
#endif
#ifndef HAVE_TDM
#define HAVE_TDM 0
#endif

union Frag {
    v16bf v;
    uint4 u[2];
    unsigned short s[16];
};

__device__ __forceinline__ unsigned short f2b(float f) {
    unsigned u = __float_as_uint(f);
    return (unsigned short)((u + 0x7FFFu + ((u >> 16) & 1u)) >> 16); // RNE
}

__device__ __forceinline__ v8f wmma_bf16(const Frag& a, const Frag& b, v8f c) {
    return __builtin_amdgcn_wmma_f32_16x16x32_bf16(
        /*neg_a=*/false, a.v, /*neg_b=*/false, b.v,
        /*c_mod=*/(short)0, c, /*reuse_a=*/false, /*reuse_b=*/false);
}

__device__ __forceinline__ float gelu_tanh(float v) {
    float c = v + 0.044715f * v * v * v;
    return 0.5f * v * (1.0f + tanhf(0.7978845608028654f * c));
}

#if HAVE_TDM
typedef unsigned int u32x4 __attribute__((ext_vector_type(4)));
typedef int          i32x4 __attribute__((ext_vector_type(4)));
typedef int          i32x8 __attribute__((ext_vector_type(8)));

// 2D tile -> LDS DMA via TDM (ISA 8.3/8.4 descriptor layout).
// tile_d0 = contiguous elements per row, tile_d1 = rows, stride in elements,
// bf16 data (data_size code 1).  pad_interval/pad_amount codes add LDS row
// padding: interval code c pads after 2<<c DWORDs, amount code a pads a+1
// DWORDs.  Tensor dims set huge so no OOB clipping occurs.
__device__ __forceinline__ void tdm_load_2d(
    unsigned lds_byte_addr, const void* gptr,
    unsigned tile_d0, unsigned tile_d1, unsigned stride_elems,
    unsigned pad_interval_code, unsigned pad_amount_code) {
    unsigned long long ga = (unsigned long long)gptr;
    u32x4 g0;
    g0.x = 1u;                                    // count=1 (valid), user mode
    g0.y = lds_byte_addr;                         // lds_addr
    g0.z = (unsigned)ga;                          // global_addr[31:0]
    g0.w = (unsigned)((ga >> 32) & 0x01FFFFFFu)   // global_addr[56:32]
         | (2u << 30);                            // type=2 ("image")
    i32x8 g1;
    g1[0] = (int)((1u << 16)                      // data_size = 2 bytes
                | (1u << 20)                      // pad_enable
                | (pad_interval_code << 22)
                | (pad_amount_code << 25));
    g1[1] = 0;                                    // no atomic barrier; td0[15:0]=0
    g1[2] = (int)0x4000;                          // tensor_dim0 = 1<<30; td1[15:0]=0
    g1[3] = (int)(0x4000u | (tile_d0 << 16));     // tensor_dim1 = 1<<30; tile_dim0
    g1[4] = (int)(tile_d1 & 0xFFFFu);             // tile_dim1; tile_dim2 = 0
    g1[5] = (int)stride_elems;                    // tensor_dim0_stride[31:0]
    g1[6] = 0;                                    // stride[47:32]; dim1_stride lo
    g1[7] = 0;
    i32x4 gz4 = {0, 0, 0, 0};
    i32x8 gz8 = {0, 0, 0, 0, 0, 0, 0, 0};
    // 6-arg form (clang-23 / therock headers): groups 0..3, aux group, cpol
    __builtin_amdgcn_tensor_load_to_lds(g0, g1, gz4, gz4, gz8, 0);
}

__device__ __forceinline__ unsigned lds_addr_of(const void* p) {
    // LDS flat addresses carry the LDS byte offset in addr[31:0] (ISA 10.2).
    return (unsigned)(unsigned long long)p;
}
#endif

// ---------------------------------------------------------------------------
// fp32 -> bf16 elementwise (8 elems / thread, 16B loads, 16B stores)
// ---------------------------------------------------------------------------
__global__ __launch_bounds__(256) void cast_f32_bf16(
    const float* __restrict__ src, unsigned short* __restrict__ dst, int n8) {
    int i = blockIdx.x * 256 + threadIdx.x;
    if (i >= n8) return;
    const float4* s4 = (const float4*)src;
    float4 a = s4[2 * i], b = s4[2 * i + 1];
    union { uint4 u; unsigned short s[8]; } o;
    o.s[0] = f2b(a.x); o.s[1] = f2b(a.y); o.s[2] = f2b(a.z); o.s[3] = f2b(a.w);
    o.s[4] = f2b(b.x); o.s[5] = f2b(b.y); o.s[6] = f2b(b.z); o.s[7] = f2b(b.w);
    ((uint4*)dst)[i] = o.u;
}

// ---------------------------------------------------------------------------
// LayerNorm: one 256-thread block per row of C=2048, bf16 output
// ---------------------------------------------------------------------------
__global__ __launch_bounds__(256) void ln_kernel(
    const float* __restrict__ x, const float* __restrict__ gamma,
    const float* __restrict__ beta, unsigned short* __restrict__ out, int C) {
    __shared__ float rs[256], rq[256];
    int row = blockIdx.x, tid = threadIdx.x;
    const float* xr = x + (size_t)row * C;
    float sum = 0.f, sq = 0.f;
    for (int i = tid * 4; i < C; i += 1024) {
        float4 v = *(const float4*)(xr + i);
        sum += v.x + v.y + v.z + v.w;
        sq  += v.x * v.x + v.y * v.y + v.z * v.z + v.w * v.w;
    }
    rs[tid] = sum; rq[tid] = sq;
    __syncthreads();
    for (int s = 128; s > 0; s >>= 1) {
        if (tid < s) { rs[tid] += rs[tid + s]; rq[tid] += rq[tid + s]; }
        __syncthreads();
    }
    float mean = rs[0] / C;
    float var  = rq[0] / C - mean * mean;
    float rstd = rsqrtf(var + 1e-5f);
    unsigned short* orow = out + (size_t)row * C;
    for (int i = tid * 4; i < C; i += 1024) {
        float4 v = *(const float4*)(xr + i);
        ushort4 o;
        o.x = f2b((v.x - mean) * rstd * gamma[i + 0] + beta[i + 0]);
        o.y = f2b((v.y - mean) * rstd * gamma[i + 1] + beta[i + 1]);
        o.z = f2b((v.z - mean) * rstd * gamma[i + 2] + beta[i + 2]);
        o.w = f2b((v.w - mean) * rstd * gamma[i + 3] + beta[i + 3]);
        *(ushort4*)(orow + i) = o;
    }
}

// ---------------------------------------------------------------------------
// bf16 GEMM  out[M,N] = A[M,K] @ B[K,N] + bias  (+ epilogue)
// Block tile 128x128, 8 waves (2 M x 4 N), wave tile 64x32 = 4x2 WMMA tiles.
// A tile: TDM DMA into double-buffered LDS [128][40] (K-contig, padded rows),
// issued one k-step ahead of the WMMA consuming it.  B tile: staged
// transposed [n][k] by all threads (TDM cannot transpose), overlapping the
// A-tile DMA.  EPI: 0 = bf16 store, 1 = GELU+bf16, 2 = +resid, f32 store.
// ---------------------------------------------------------------------------
template <int EPI>
__global__ __launch_bounds__(256) void gemm_bf16(
    const unsigned short* __restrict__ A, const unsigned short* __restrict__ Bw,
    const float* __restrict__ bias, const float* __restrict__ resid,
    void* __restrict__ out, int M, int N, int K) {
    constexpr int LDA = 40, LDB = 40;
    __shared__ unsigned short a_lds[2 * 128 * LDA];
    __shared__ unsigned short bt_lds[128 * LDB];

    int n0 = blockIdx.x * 128, m0 = blockIdx.y * 128;
    int tid = threadIdx.x;
    int wave = tid >> 5, lane = tid & 31;
    int half = lane >> 4, l16 = lane & 15;
    int wm = wave >> 2, wn = wave & 3;          // 2 x 4 wave grid
    int wrow = m0 + wm * 64, wcol = n0 + wn * 32;

    v8f acc[4][2];
    v8f zed = {0.f, 0.f, 0.f, 0.f, 0.f, 0.f, 0.f, 0.f};
#pragma unroll
    for (int i = 0; i < 4; ++i)
#pragma unroll
        for (int j = 0; j < 2; ++j) acc[i][j] = zed;

    const int nk = K / 32;
#if HAVE_TDM
    unsigned aBase = lds_addr_of(a_lds);
    if (wave == 0)   // prefetch tile 0 into buffer 0
        tdm_load_2d(aBase, A + (size_t)m0 * K, 32, 128, (unsigned)K, 3, 3);
#endif

    for (int ks = 0; ks < nk; ++ks) {
        const int k0 = ks * 32;
        __syncthreads();   // previous iteration's fragment reads done
        // stage B transposed: 32 k x 128 n (overlaps A-tile DMA)
#pragma unroll
        for (int l = 0; l < 2; ++l) {
            int c = tid + l * 256;
            int kr = c >> 4, col8 = (c & 15) * 8;
            union { uint4 u; unsigned short s[8]; } d;
            d.u = *(const uint4*)(Bw + (size_t)(k0 + kr) * N + n0 + col8);
#pragma unroll
            for (int j = 0; j < 8; ++j) bt_lds[(col8 + j) * LDB + kr] = d.s[j];
        }
#if HAVE_TDM
        if (wave == 0) {
            if (ks + 1 < nk) {
                tdm_load_2d(aBase + (unsigned)(((ks + 1) & 1) * 128 * LDA * 2),
                            A + (size_t)m0 * K + (k0 + 32), 32, 128,
                            (unsigned)K, 3, 3);
                __builtin_amdgcn_s_wait_tensorcnt(1);  // tile ks landed
            } else {
                __builtin_amdgcn_s_wait_tensorcnt(0);
            }
        }
        const unsigned short* acur = a_lds + (ks & 1) * 128 * LDA;
#else
        // manual A stage: 128 rows x 32 k, 512 16B-chunks, 2 per thread
#pragma unroll
        for (int l = 0; l < 2; ++l) {
            int c = tid + l * 256;
            int row = c >> 2, col8 = (c & 3) * 8;
            uint4 d = *(const uint4*)(A + (size_t)(m0 + row) * K + k0 + col8);
            *(uint4*)(a_lds + row * LDA + col8) = d;
        }
        const unsigned short* acur = a_lds;
#endif
        __syncthreads();

        Frag af[4], bf[2];
#pragma unroll
        for (int i = 0; i < 4; ++i) {
            int row = wm * 64 + i * 16 + l16;
            af[i].u[0] = *(const uint4*)(acur + row * LDA + half * 8);
            af[i].u[1] = *(const uint4*)(acur + row * LDA + 16 + half * 8);
        }
#pragma unroll
        for (int j = 0; j < 2; ++j) {
            int col = wn * 32 + j * 16 + l16;
            bf[j].u[0] = *(const uint4*)(bt_lds + col * LDB + half * 8);
            bf[j].u[1] = *(const uint4*)(bt_lds + col * LDB + 16 + half * 8);
        }
#pragma unroll
        for (int i = 0; i < 4; ++i)
#pragma unroll
            for (int j = 0; j < 2; ++j)
                acc[i][j] = wmma_bf16(af[i], bf[j], acc[i][j]);
    }

    // epilogue: C/D frag row = r + 8*half, col = lane&15
#pragma unroll
    for (int i = 0; i < 4; ++i) {
#pragma unroll
        for (int j = 0; j < 2; ++j) {
            int col = wcol + j * 16 + l16;
            float bv = bias[col];
#pragma unroll
            for (int r = 0; r < 8; ++r) {
                int row = wrow + i * 16 + r + half * 8;
                size_t off = (size_t)row * N + col;
                float v = acc[i][j][r] + bv;
                if constexpr (EPI == 0) {
                    ((unsigned short*)out)[off] = f2b(v);
                } else if constexpr (EPI == 1) {
                    ((unsigned short*)out)[off] = f2b(gelu_tanh(v));
                } else {
                    ((float*)out)[off] = v + resid[off];
                }
            }
        }
    }
}

// ---------------------------------------------------------------------------
// Flash attention, causal, D=128, bf16 QKV in [B*T, 3C] (Q|K|V per token).
// Block: 128 threads = 4 waves; block owns 64 q-rows of one (b,h); each wave
// owns 16 q-rows.  Q and K tiles DMA'd by the TDM (padded rows -> LDS stride
// 136); V staged transposed by all threads between TDM issue and wait.
// Online softmax in f32 on the C-frag layout; row statistics reduced across
// the owning 16-lane half with shfl_xor width 16.
// ---------------------------------------------------------------------------
__global__ __launch_bounds__(128) void flash_attn(
    const unsigned short* __restrict__ qkv, unsigned short* __restrict__ obuf,
    int T, int C3, int Cdim) {
    constexpr int D = 128;
    __shared__ unsigned short qtile[64 * 136];
    __shared__ unsigned short ktile[64 * 136];
    __shared__ unsigned short vtt[128 * 72];     // [d][key]
    __shared__ unsigned short pbuf[64 * 72];     // per-wave 16x64 P tiles

    int bh = blockIdx.y;
    int b = bh >> 4, h = bh & 15;
    int qblk = blockIdx.x, q0 = qblk * 64;
    size_t base = (size_t)b * T * C3;

    int tid = threadIdx.x;
    int wave = tid >> 5, lane = tid & 31;
    int half = lane >> 4, l16 = lane & 15;

    // stage Q tile (64 x 128)
#if HAVE_TDM
    if (wave == 0) {
        tdm_load_2d(lds_addr_of(qtile), qkv + base + (size_t)q0 * C3 + h * D,
                    128, 64, (unsigned)C3, 5, 3);
        __builtin_amdgcn_s_wait_tensorcnt(0);
    }
#else
    for (int c = tid; c < 1024; c += 128) {
        int row = c >> 4, col8 = (c & 15) * 8;
        uint4 d = *(const uint4*)(qkv + base + (size_t)(q0 + row) * C3 + h * D + col8);
        *(uint4*)(qtile + row * 136 + col8) = d;
    }
#endif
    __syncthreads();

    Frag qf[4];
    int qrow_l = wave * 16 + l16;
#pragma unroll
    for (int c = 0; c < 4; ++c) {
        qf[c].u[0] = *(const uint4*)(qtile + qrow_l * 136 + c * 32 + half * 8);
        qf[c].u[1] = *(const uint4*)(qtile + qrow_l * 136 + c * 32 + 16 + half * 8);
    }

    v8f Oacc[8];
    v8f zed = {0.f, 0.f, 0.f, 0.f, 0.f, 0.f, 0.f, 0.f};
#pragma unroll
    for (int nb = 0; nb < 8; ++nb) Oacc[nb] = zed;
    float ms[8], ls[8];
#pragma unroll
    for (int r = 0; r < 8; ++r) { ms[r] = -1e30f; ls[r] = 0.f; }
    const float scale = 0.08838834764831845f;   // 1/sqrt(128)
    const float NEGBIG = -3.0e38f;

    for (int jt = 0; jt <= qblk; ++jt) {
        int kb = jt * 64;
        __syncthreads();   // previous KV-tile reads done
#if HAVE_TDM
        if (wave == 0)     // K tile DMA, overlapped with V staging below
            tdm_load_2d(lds_addr_of(ktile),
                        qkv + base + (size_t)kb * C3 + Cdim + h * D,
                        128, 64, (unsigned)C3, 5, 3);
        for (int c = tid; c < 1024; c += 128) {
            int row = c >> 4, col8 = (c & 15) * 8;
            union { uint4 u; unsigned short s[8]; } dv;
            dv.u = *(const uint4*)(qkv + base + (size_t)(kb + row) * C3 + 2 * Cdim + h * D + col8);
#pragma unroll
            for (int j = 0; j < 8; ++j) vtt[(col8 + j) * 72 + row] = dv.s[j];
        }
        if (wave == 0) __builtin_amdgcn_s_wait_tensorcnt(0);
#else
        for (int c = tid; c < 1024; c += 128) {
            int row = c >> 4, col8 = (c & 15) * 8;
            uint4 dk = *(const uint4*)(qkv + base + (size_t)(kb + row) * C3 + Cdim + h * D + col8);
            *(uint4*)(ktile + row * 136 + col8) = dk;
            union { uint4 u; unsigned short s[8]; } dv;
            dv.u = *(const uint4*)(qkv + base + (size_t)(kb + row) * C3 + 2 * Cdim + h * D + col8);
#pragma unroll
            for (int j = 0; j < 8; ++j) vtt[(col8 + j) * 72 + row] = dv.s[j];
        }
#endif
        __syncthreads();

        bool diag = (jt == qblk);
#pragma unroll 1
        for (int j2 = 0; j2 < 4; ++j2) {
            // S = Q @ K^T over D in 4 chunks of 32
            v8f s = zed;
            int key_l = j2 * 16 + l16;
#pragma unroll
            for (int c = 0; c < 4; ++c) {
                Frag kf;
                kf.u[0] = *(const uint4*)(ktile + key_l * 136 + c * 32 + half * 8);
                kf.u[1] = *(const uint4*)(ktile + key_l * 136 + c * 32 + 16 + half * 8);
                s = wmma_bf16(qf[c], kf, s);
            }
            int kj = kb + key_l;
#pragma unroll
            for (int r = 0; r < 8; ++r) {
                float sv = s[r] * scale;
                if (diag) {
                    int qi = q0 + wave * 16 + r + half * 8;
                    if (kj > qi) sv = NEGBIG;
                }
                s[r] = sv;
            }
            // online softmax update (per 16-lane half)
#pragma unroll
            for (int r = 0; r < 8; ++r) {
                float v = s[r];
                float rmax = v;
                rmax = fmaxf(rmax, __shfl_xor(rmax, 1, 16));
                rmax = fmaxf(rmax, __shfl_xor(rmax, 2, 16));
                rmax = fmaxf(rmax, __shfl_xor(rmax, 4, 16));
                rmax = fmaxf(rmax, __shfl_xor(rmax, 8, 16));
                float mnew = fmaxf(fmaxf(ms[r], rmax), -1e30f);
                float p = __expf(v - mnew);
                float rsum = p;
                rsum += __shfl_xor(rsum, 1, 16);
                rsum += __shfl_xor(rsum, 2, 16);
                rsum += __shfl_xor(rsum, 4, 16);
                rsum += __shfl_xor(rsum, 8, 16);
                float corr = __expf(ms[r] - mnew);
                ls[r] = ls[r] * corr + rsum;
                ms[r] = mnew;
#pragma unroll
                for (int nb = 0; nb < 8; ++nb) Oacc[nb][r] *= corr;
                pbuf[(wave * 16 + r + half * 8) * 72 + j2 * 16 + l16] = f2b(p);
            }
        }
        // O += P(16x64) @ V(64x128): A = P (per-wave pbuf), B = V^T tile
#pragma unroll
        for (int kc = 0; kc < 2; ++kc) {
            Frag pf;
            pf.u[0] = *(const uint4*)(pbuf + (wave * 16 + l16) * 72 + kc * 32 + half * 8);
            pf.u[1] = *(const uint4*)(pbuf + (wave * 16 + l16) * 72 + kc * 32 + 16 + half * 8);
#pragma unroll
            for (int nb = 0; nb < 8; ++nb) {
                Frag vf;
                int dcol = nb * 16 + l16;
                vf.u[0] = *(const uint4*)(vtt + dcol * 72 + kc * 32 + half * 8);
                vf.u[1] = *(const uint4*)(vtt + dcol * 72 + kc * 32 + 16 + half * 8);
                Oacc[nb] = wmma_bf16(pf, vf, Oacc[nb]);
            }
        }
    }

    // normalize and store O as bf16 [B*T, C], col = h*128 + d
    float inv[8];
#pragma unroll
    for (int r = 0; r < 8; ++r) inv[r] = 1.0f / ls[r];
    size_t obase = (size_t)b * T * Cdim;
#pragma unroll
    for (int nb = 0; nb < 8; ++nb) {
#pragma unroll
        for (int r = 0; r < 8; ++r) {
            int qrow = q0 + wave * 16 + r + half * 8;
            obuf[obase + (size_t)qrow * Cdim + h * D + nb * 16 + l16] =
                f2b(Oacc[nb][r] * inv[r]);
        }
    }
}

// ---------------------------------------------------------------------------
extern "C" void kernel_launch(void* const* d_in, const int* in_sizes, int n_in,
                              void* d_out, int out_size, void* d_ws, size_t ws_size,
                              hipStream_t stream) {
    const int B = 2, T = 2048, C = 2048, Dff = 8192;
    const int M = B * T, C3 = 3 * C;

    const float* x        = (const float*)d_in[0];
    const float* ln1_s    = (const float*)d_in[1];
    const float* ln1_b    = (const float*)d_in[2];
    const float* c_attn_w = (const float*)d_in[3];
    const float* c_attn_b = (const float*)d_in[4];
    const float* aproj_w  = (const float*)d_in[5];
    const float* aproj_b  = (const float*)d_in[6];
    const float* ln2_s    = (const float*)d_in[7];
    const float* ln2_b    = (const float*)d_in[8];
    const float* fc_w     = (const float*)d_in[9];
    const float* fc_b     = (const float*)d_in[10];
    const float* mlp_w    = (const float*)d_in[11];
    const float* mlp_b    = (const float*)d_in[12];

    char* ws = (char*)d_ws;
    unsigned short* wqkv = (unsigned short*)(ws + 0);          // 25.2 MB
    unsigned short* wap  = (unsigned short*)(ws + 25165824);   //  8.4 MB
    unsigned short* wfc  = (unsigned short*)(ws + 33554432);   // 33.6 MB
    unsigned short* wmlp = (unsigned short*)(ws + 67108864);   // 33.6 MB
    unsigned short* lnb  = (unsigned short*)(ws + 100663296);  // 16.8 MB (LN1 & LN2)
    float*          x1   = (float*)(ws + 117440512);           // 33.6 MB
    unsigned short* qkvb = (unsigned short*)(ws + 150994944);  // 50.3 MB
    unsigned short* ob   = (unsigned short*)(ws + 201326592);  // 16.8 MB (end ~218 MB)
    unsigned short* hb   = qkvb;  // GELU buffer aliases dead qkv+obuf (67.1 MB)

    // one-shot fp32 -> bf16 weight conversion
    cast_f32_bf16<<<dim3((C * C3 / 8 + 255) / 256), 256, 0, stream>>>(c_attn_w, wqkv, C * C3 / 8);
    cast_f32_bf16<<<dim3((C * C  / 8 + 255) / 256), 256, 0, stream>>>(aproj_w,  wap,  C * C  / 8);
    cast_f32_bf16<<<dim3((C * Dff/ 8 + 255) / 256), 256, 0, stream>>>(fc_w,     wfc,  C * Dff/ 8);
    cast_f32_bf16<<<dim3((Dff * C/ 8 + 255) / 256), 256, 0, stream>>>(mlp_w,    wmlp, Dff * C/ 8);

    // ---- attention branch ----
    ln_kernel<<<dim3(M), 256, 0, stream>>>(x, ln1_s, ln1_b, lnb, C);
    gemm_bf16<0><<<dim3(C3 / 128, M / 128), 256, 0, stream>>>(
        lnb, wqkv, c_attn_b, nullptr, qkvb, M, C3, C);
    flash_attn<<<dim3(T / 64, B * 16), 128, 0, stream>>>(qkvb, ob, T, C3, C);
    gemm_bf16<2><<<dim3(C / 128, M / 128), 256, 0, stream>>>(
        ob, wap, aproj_b, x, x1, M, C, C);

    // ---- MLP branch ----
    ln_kernel<<<dim3(M), 256, 0, stream>>>(x1, ln2_s, ln2_b, lnb, C);
    gemm_bf16<1><<<dim3(Dff / 128, M / 128), 256, 0, stream>>>(
        lnb, wfc, fc_b, nullptr, hb, M, Dff, C);
    gemm_bf16<2><<<dim3(C / 128, M / 128), 256, 0, stream>>>(
        hb, wmlp, mlp_b, x1, (float*)d_out, M, C, Dff);
}
